// Block_9405978378750
// MI455X (gfx1250) — matrix operand order repaired and checked by reading, not compiled, not run on hardware
//
#include <hip/hip_runtime.h>

// ---------------------------------------------------------------------------
// Transformer block (pre-norm attention + MLP) for MI455X / gfx1250.
// All GEMMs run through v_wmma_f32_16x16x32_bf16 (f32 accumulate).
// GEMM staging uses CDNA5 async global->LDS DMA (ASYNCcnt) + double buffering.
// B=2, T=2048, D=1024, H=16, hd=64, DFF=4096.
// ---------------------------------------------------------------------------

typedef __attribute__((ext_vector_type(16))) __bf16 v16bf;
typedef __attribute__((ext_vector_type(8)))  float  v8f;

typedef unsigned short u16;

union Frag {
    uint4 u[2];   // 32 bytes
    v16bf v;      // 16 x bf16
};

__device__ __forceinline__ u16 f2bf(float f) {
    unsigned int u = __float_as_uint(f);
    unsigned int r = u + 0x7fffu + ((u >> 16) & 1u);   // round-to-nearest-even
    return (u16)(r >> 16);
}

__device__ __forceinline__ v8f wmma_bf16(const Frag& a, const Frag& b, v8f c) {
    return __builtin_amdgcn_wmma_f32_16x16x32_bf16(
        /*neg_a=*/false, a.v, /*neg_b=*/false, b.v,
        /*c_mod=*/(short)0, c, /*reuse_a=*/false, /*reuse_b=*/false);
}

// Async global -> LDS 16B DMA (CDNA5, tracked by ASYNCcnt).
__device__ __forceinline__ void async_ld_b128(unsigned lds_off, const void* gptr) {
    asm volatile("global_load_async_to_lds_b128 %0, %1, off"
                 :: "v"(lds_off), "v"((unsigned long long)gptr) : "memory");
}
__device__ __forceinline__ void wait_async0() {
    asm volatile("s_wait_asynccnt 0" ::: "memory");
}

// ----------------------------- constants -----------------------------------
constexpr int Tlen = 2048;
constexpr int Dm   = 1024;
constexpr int Hh   = 16;
constexpr int HD   = 64;
constexpr int DFF  = 4096;
constexpr int BT   = 2 * Tlen;   // 4096 rows

// ---------------------------------------------------------------------------
// Weight convert + transpose: Wt[n][k] = bf16(W[k][n]).   (N-major for WMMA B)
// ---------------------------------------------------------------------------
__global__ void convw(const float* __restrict__ W, u16* __restrict__ Wt,
                      int K, int N) {
    long id = (long)blockIdx.x * 256 + threadIdx.x;
    int k = (int)(id / N);
    int n = (int)(id % N);
    Wt[(size_t)n * K + k] = f2bf(W[id]);
}

// ---------------------------------------------------------------------------
// LayerNorm: one block per row of [BT, 1024], fp32 in -> bf16 out.
// ---------------------------------------------------------------------------
__global__ __launch_bounds__(256) void ln_kernel(const float* __restrict__ x,
                                                 const float* __restrict__ g,
                                                 const float* __restrict__ bb,
                                                 u16* __restrict__ out) {
    int row  = blockIdx.x;
    int tid  = threadIdx.x;
    int lane = tid & 31;
    int wave = tid >> 5;
    const float* xr = x + (size_t)row * Dm;
    float4 v = *(const float4*)(xr + tid * 4);
    float s  = v.x + v.y + v.z + v.w;
    float ss = v.x * v.x + v.y * v.y + v.z * v.z + v.w * v.w;
#pragma unroll
    for (int off = 16; off > 0; off >>= 1) {
        s  += __shfl_xor(s,  off, 32);
        ss += __shfl_xor(ss, off, 32);
    }
    __shared__ float sm[8], sm2[8];
    if (lane == 0) { sm[wave] = s; sm2[wave] = ss; }
    __syncthreads();
    if (tid == 0) {
        float t = 0.f, t2 = 0.f;
#pragma unroll
        for (int i = 0; i < 8; i++) { t += sm[i]; t2 += sm2[i]; }
        float mean = t * (1.0f / Dm);
        float var  = t2 * (1.0f / Dm) - mean * mean;
        sm[0]  = mean;
        sm2[0] = rsqrtf(var + 1e-5f);
    }
    __syncthreads();
    float mean = sm[0], rstd = sm2[0];
    int c = tid * 4;
    float4 gv = *(const float4*)(g + c);
    float4 bv = *(const float4*)(bb + c);
    u16 o0 = f2bf((v.x - mean) * rstd * gv.x + bv.x);
    u16 o1 = f2bf((v.y - mean) * rstd * gv.y + bv.y);
    u16 o2 = f2bf((v.z - mean) * rstd * gv.z + bv.z);
    u16 o3 = f2bf((v.w - mean) * rstd * gv.w + bv.w);
    ushort4 pk = make_ushort4(o0, o1, o2, o3);
    *(ushort4*)(out + (size_t)row * Dm + c) = pk;
}

// ---------------------------------------------------------------------------
// Tiled WMMA GEMM:  C[M,N] = A[M,K](bf16, row-major) * Bt[N,K](bf16, N-major)
// Block: 256 threads = 8 waves, tile 128x128, BK=32, double-buffered LDS
// filled by async global->LDS DMA. Wave grid 2(M) x 4(N): each wave owns
// 64x32 = 4x2 C tiles of 16x16.
// MODE 1: write bf16 qkv + scatter transposed V to vt
// MODE 2: fp32 out = acc + residual
// MODE 3: bf16 out = gelu(acc)
// MODE 4: fp32 out = acc + residual
// ---------------------------------------------------------------------------
template <int MODE>
__global__ __launch_bounds__(256) void gemm_kernel(
    const u16* __restrict__ A, const u16* __restrict__ Bt,
    int M, int N, int K,
    u16* __restrict__ Cb, float* __restrict__ Cf,
    const float* __restrict__ res, u16* __restrict__ vt) {

    __shared__ uint4 As[2][128][4];   // 128 rows x 32 bf16, double buffered
    __shared__ uint4 Bs[2][128][4];   // 128 cols(N) x 32 bf16 (K-contiguous)

    const int tid  = threadIdx.x;
    const int lane = tid & 31;
    const int wave = tid >> 5;
    const int half = lane >> 4;
    const int l15  = lane & 15;
    const int wm   = wave & 1;   // 0..1
    const int wn   = wave >> 1;  // 0..3
    const int bm   = blockIdx.y;
    const int bn   = blockIdx.x;

    const int r0  = tid >> 1;        // row this thread stages (2 x 16B each)
    const int c40 = (tid & 1) * 2;   // starting 16B chunk

    auto stage = [&](int kc, int bufi) {
#pragma unroll
        for (int i = 0; i < 2; i++) {
            int c4 = c40 + i;
            async_ld_b128((unsigned)(uintptr_t)&As[bufi][r0][c4],
                          A  + (size_t)(bm * 128 + r0) * K + kc * 32 + c4 * 8);
            async_ld_b128((unsigned)(uintptr_t)&Bs[bufi][r0][c4],
                          Bt + (size_t)(bn * 128 + r0) * K + kc * 32 + c4 * 8);
        }
    };

    v8f acc[4][2] = {};

    const int nk = K >> 5;
    stage(0, 0);
    wait_async0();
    __syncthreads();

    for (int kc = 0; kc < nk; kc++) {
        const int cur = kc & 1;
        if (kc + 1 < nk) stage(kc + 1, cur ^ 1);   // overlap DMA with math

        Frag af[4], bfr[2];
#pragma unroll
        for (int mt = 0; mt < 4; mt++) {
            int r = wm * 64 + mt * 16 + l15;
            af[mt].u[0] = As[cur][r][half];        // K = 0..7   (or 8..15)
            af[mt].u[1] = As[cur][r][2 + half];    // K = 16..23 (or 24..31)
        }
#pragma unroll
        for (int nt = 0; nt < 2; nt++) {
            int r = wn * 32 + nt * 16 + l15;
            bfr[nt].u[0] = Bs[cur][r][half];          // K = 0..7   (or 8..15)
            bfr[nt].u[1] = Bs[cur][r][2 + half];      // K = 16..23 (or 24..31)
        }
#pragma unroll
        for (int mt = 0; mt < 4; mt++)
#pragma unroll
            for (int nt = 0; nt < 2; nt++)
                acc[mt][nt] = wmma_bf16(af[mt], bfr[nt], acc[mt][nt]);

        wait_async0();
        __syncthreads();
    }

    // ---------------- epilogue ----------------
#pragma unroll
    for (int mt = 0; mt < 4; mt++) {
#pragma unroll
        for (int nt = 0; nt < 2; nt++) {
#pragma unroll
            for (int e = 0; e < 8; e++) {
                int gm = bm * 128 + wm * 64 + mt * 16 + e + half * 8;
                int gn = bn * 128 + wn * 32 + nt * 16 + l15;
                float v = acc[mt][nt][e];
                if constexpr (MODE == 1) {
                    Cb[(size_t)gm * N + gn] = f2bf(v);
                    if (gn >= 2 * Dm) {              // V part -> transposed layout
                        int head = (gn - 2 * Dm) >> 6;
                        int hdi  = (gn - 2 * Dm) & 63;
                        int b    = gm >> 11;         // / Tlen
                        int t    = gm & (Tlen - 1);
                        vt[((size_t)((b * Hh + head) * HD + hdi)) * Tlen + t] = f2bf(v);
                    }
                } else if constexpr (MODE == 2) {
                    Cf[(size_t)gm * N + gn] = v + res[(size_t)gm * N + gn];
                } else if constexpr (MODE == 3) {
                    float gl = 0.5f * v * (1.0f + erff(v * 0.70710678118654752f));
                    Cb[(size_t)gm * N + gn] = f2bf(gl);
                } else { // MODE == 4
                    Cf[(size_t)gm * N + gn] = v + res[(size_t)gm * N + gn];
                }
            }
        }
    }
}

// ---------------------------------------------------------------------------
// Flash attention. One 16-row query strip per wave (fully wave-local; no
// block barriers so causal loop-length divergence between waves is safe).
// S = Q*K^T via WMMA, online softmax, P staged through per-wave LDS
// (C-layout -> A-layout), then P*V via WMMA against transposed V (vt).
// Only the final 32-key chunk straddles the causal diagonal, so masking is
// a wave-uniform branch taken once per strip.
// ---------------------------------------------------------------------------
__global__ __launch_bounds__(256) void attn_kernel(const u16* __restrict__ qkv,
                                                   const u16* __restrict__ vt,
                                                   u16* __restrict__ y) {
    const int lane = threadIdx.x & 31;
    const int wave = threadIdx.x >> 5;
    const int gw   = blockIdx.x * 8 + wave;     // 0..4095
    const int qs   = gw & 127;                  // query strip
    const int h    = (gw >> 7) & 15;
    const int b    = gw >> 11;
    const int q0   = qs * 16;
    const int half = lane >> 4;
    const int l15  = lane & 15;

    __shared__ __align__(16) u16 Pst[8][16][32];

    const float scale = 0.125f;  // 1/sqrt(64)

    // Q fragments (A-layout), 16 rows x 64 (two K-chunks of 32)
    Frag qa[2];
    {
        const u16* qrow = qkv + (size_t)(b * Tlen + q0 + l15) * (3 * Dm) + h * HD;
#pragma unroll
        for (int c = 0; c < 2; c++) {
            int k0 = c * 32 + half * 16;
            qa[c].u[0] = *(const uint4*)(qrow + k0);
            qa[c].u[1] = *(const uint4*)(qrow + k0 + 8);
        }
    }

    v8f o[4] = {};
    float mrow[8], lrow[8];
#pragma unroll
    for (int e = 0; e < 8; e++) { mrow[e] = -__builtin_inff(); lrow[e] = 0.0f; }

    const int kend = q0 + 16;
    const int nch  = (kend + 31) >> 5;
    for (int ch = 0; ch < nch; ch++) {
        const int kc = ch * 32;
        const bool needmask = (ch == nch - 1);   // uniform across the wave

        // ---- S = Q * K^T for 32 keys (two 16-key tiles) ----
        v8f s0 = {}, s1 = {};
#pragma unroll
        for (int c = 0; c < 2; c++) {        // hd chunk (K dim of WMMA)
            Frag kb0, kb1;
            int hb = c * 32 + half * 16;
            const u16* kr0 = qkv + (size_t)(b * Tlen + kc + l15)      * (3 * Dm) + Dm + h * HD + hb;
            const u16* kr1 = qkv + (size_t)(b * Tlen + kc + 16 + l15) * (3 * Dm) + Dm + h * HD + hb;
            kb0.u[0] = *(const uint4*)kr0; kb0.u[1] = *(const uint4*)(kr0 + 8);
            kb1.u[0] = *(const uint4*)kr1; kb1.u[1] = *(const uint4*)(kr1 + 8);
            s0 = wmma_bf16(qa[c], kb0, s0);
            s1 = wmma_bf16(qa[c], kb1, s1);
        }

        // ---- online softmax update ----
        u16 pb0[8], pb1[8];
#pragma unroll
        for (int e = 0; e < 8; e++) {
            int row = q0 + e + half * 8;
            float x0 = s0[e] * scale;
            float x1 = s1[e] * scale;
            if (needmask) {
                if (kc + l15      > row) x0 = -__builtin_inff();
                if (kc + 16 + l15 > row) x1 = -__builtin_inff();
            }
            float cm = fmaxf(x0, x1);
#pragma unroll
            for (int off = 1; off < 16; off <<= 1)
                cm = fmaxf(cm, __shfl_xor(cm, off, 32));
            float mnew  = fmaxf(mrow[e], cm);
            float alpha = __expf(mrow[e] - mnew);
            float p0 = __expf(x0 - mnew);
            float p1 = __expf(x1 - mnew);
            float ps = p0 + p1;
#pragma unroll
            for (int off = 1; off < 16; off <<= 1)
                ps += __shfl_xor(ps, off, 32);
            lrow[e] = lrow[e] * alpha + ps;
            mrow[e] = mnew;
#pragma unroll
            for (int n = 0; n < 4; n++) o[n][e] *= alpha;
            pb0[e] = f2bf(p0);
            pb1[e] = f2bf(p1);
        }

        // ---- stage P (C-layout -> row-major LDS), wave-local ----
#pragma unroll
        for (int e = 0; e < 8; e++) {
            Pst[wave][e + half * 8][l15]      = pb0[e];
            Pst[wave][e + half * 8][16 + l15] = pb1[e];
        }
        __builtin_amdgcn_wave_barrier();

        Frag pa;
        {
            const u16* pr = &Pst[wave][l15][0];
            int k0 = half * 16;
            pa.u[0] = *(const uint4*)(pr + k0);
            pa.u[1] = *(const uint4*)(pr + k0 + 8);
        }
        __builtin_amdgcn_wave_barrier();

        // ---- O += P * V  (V transposed: vt[b,h,hd,T]) ----
        const u16* vbase = vt + (size_t)((b * Hh + h) * HD) * Tlen;
#pragma unroll
        for (int n = 0; n < 4; n++) {
            Frag vb;
            const u16* vr = vbase + (size_t)(n * 16 + l15) * Tlen + kc + half * 16;
            vb.u[0] = *(const uint4*)vr;
            vb.u[1] = *(const uint4*)(vr + 8);
            o[n] = wmma_bf16(pa, vb, o[n]);
        }
    }

    // ---- normalize + write y (bf16, [BT, D]) ----
#pragma unroll
    for (int e = 0; e < 8; e++) {
        float inv = 1.0f / lrow[e];
        int row = b * Tlen + q0 + e + half * 8;
#pragma unroll
        for (int n = 0; n < 4; n++) {
            y[(size_t)row * Dm + h * HD + n * 16 + l15] = f2bf(o[n][e] * inv);
        }
    }
}

// ---------------------------------------------------------------------------
// Host side
// ---------------------------------------------------------------------------
extern "C" void kernel_launch(void* const* d_in, const int* in_sizes, int n_in,
                              void* d_out, int out_size, void* d_ws, size_t ws_size,
                              hipStream_t stream) {
    const float* x      = (const float*)d_in[0];
    const float* W_qkv  = (const float*)d_in[1];
    const float* W_proj = (const float*)d_in[2];
    const float* W_fc1  = (const float*)d_in[3];
    const float* W_fc2  = (const float*)d_in[4];
    const float* ln1_g  = (const float*)d_in[5];
    const float* ln1_b  = (const float*)d_in[6];
    const float* ln2_g  = (const float*)d_in[7];
    const float* ln2_b  = (const float*)d_in[8];
    float* out = (float*)d_out;

    char* ws = (char*)d_ws;
    size_t off = 0;
    auto alloc = [&](size_t bytes) {
        char* p = ws + off;
        off += (bytes + 255) & ~(size_t)255;
        return p;
    };
    u16*   wqkv_t  = (u16*)  alloc((size_t)(3 * Dm) * Dm * 2);   // [3072][1024]
    u16*   wproj_t = (u16*)  alloc((size_t)Dm * Dm * 2);         // [1024][1024]
    u16*   wfc1_t  = (u16*)  alloc((size_t)DFF * Dm * 2);        // [4096][1024]
    u16*   wfc2_t  = (u16*)  alloc((size_t)Dm * DFF * 2);        // [1024][4096]
    u16*   hbuf    = (u16*)  alloc((size_t)BT * Dm * 2);         // ln out (reused)
    u16*   qkv     = (u16*)  alloc((size_t)BT * 3 * Dm * 2);     // [4096][3072]
    u16*   vtb     = (u16*)  alloc((size_t)2 * Hh * HD * Tlen * 2);
    u16*   ybuf    = (u16*)  alloc((size_t)BT * Dm * 2);
    u16*   ffh     = (u16*)  alloc((size_t)BT * DFF * 2);
    float* x2      = (float*)alloc((size_t)BT * Dm * 4);

    // 1) convert + transpose weights to bf16 (N-major)
    convw<<<(Dm * 3 * Dm) / 256, 256, 0, stream>>>(W_qkv,  wqkv_t,  Dm,  3 * Dm);
    convw<<<(Dm * Dm)     / 256, 256, 0, stream>>>(W_proj, wproj_t, Dm,  Dm);
    convw<<<(Dm * DFF)    / 256, 256, 0, stream>>>(W_fc1,  wfc1_t,  Dm,  DFF);
    convw<<<(DFF * Dm)    / 256, 256, 0, stream>>>(W_fc2,  wfc2_t,  DFF, Dm);

    // 2) LN1
    ln_kernel<<<BT, 256, 0, stream>>>(x, ln1_g, ln1_b, hbuf);

    // 3) QKV GEMM (+ transposed V scatter)
    gemm_kernel<1><<<dim3((3 * Dm) / 128, BT / 128), 256, 0, stream>>>(
        hbuf, wqkv_t, BT, 3 * Dm, Dm, qkv, nullptr, nullptr, vtb);

    // 4) attention
    attn_kernel<<<(2 * Hh * (Tlen / 16)) / 8, 256, 0, stream>>>(qkv, vtb, ybuf);

    // 5) projection + residual -> x2 (fp32)
    gemm_kernel<2><<<dim3(Dm / 128, BT / 128), 256, 0, stream>>>(
        ybuf, wproj_t, BT, Dm, Dm, nullptr, x2, x, nullptr);

    // 6) LN2
    ln_kernel<<<BT, 256, 0, stream>>>(x2, ln2_g, ln2_b, hbuf);

    // 7) fc1 + GELU -> ffh (bf16)
    gemm_kernel<3><<<dim3(DFF / 128, BT / 128), 256, 0, stream>>>(
        hbuf, wfc1_t, BT, DFF, Dm, ffh, nullptr, nullptr, nullptr);

    // 8) fc2 + residual -> out (fp32)
    gemm_kernel<4><<<dim3(Dm / 128, BT / 128), 256, 0, stream>>>(
        ffh, wfc2_t, BT, Dm, DFF, nullptr, out, x2, nullptr);

    (void)in_sizes; (void)n_in; (void)out_size; (void)ws_size;
}
